// QuantumMarioNet_57921928954276
// MI455X (gfx1250) — matrix operand order repaired and checked by reading, not compiled, not run on hardware
//
#include <hip/hip_runtime.h>
#include <hip/hip_bf16.h>
#include <math.h>

typedef __attribute__((ext_vector_type(16))) __bf16        v16bf;
typedef __attribute__((ext_vector_type(8)))  float         v8f;
typedef __attribute__((ext_vector_type(4)))  unsigned int  u32x4;
typedef __attribute__((ext_vector_type(8)))  unsigned int  u32x8;

#define NQ      8
#define QDEPTH  2
#define CHIPS   100
#define BATCH   128
#define IN_DIM  (4*84*84)
#define HID     256
#define HID2    64
#define ACTIONS 7

#define GWAVES  4           // waves per GEMM workgroup
#define KCHUNK  32          // K elements staged per TDM load

// ---------------------------------------------------------------------------
// Tensor Data Mover: issue a 2-D tile load (tile_d0 x tile_d1 f32, row stride
// stride0 elements) from global memory into LDS at byte offset lds_off.
// D# layout per CDNA5 ISA (group0 128b, group1 256b);
// groups 2/3 omitted (2-D tensor -> NULL).  Tracked by TENSORcnt.
// ---------------------------------------------------------------------------
__device__ __forceinline__ void tdm_load_2d(const float* gaddr, unsigned lds_off,
                                            unsigned tile_d0, unsigned tile_d1,
                                            unsigned long long stride0)
{
  unsigned long long ga = (unsigned long long)(uintptr_t)gaddr;
  u32x4 g0;
  g0[0] = 1u;                                        // count=1, user-mode, no gather
  g0[1] = lds_off;                                   // lds_addr (bytes)
  g0[2] = (unsigned)(ga & 0xFFFFFFFFu);              // global_addr[31:0]
  g0[3] = (unsigned)((ga >> 32) & 0x01FFFFFFu)       // global_addr[56:32]
        | 0x80000000u;                               // type=2 ("image") bits 127:126

  unsigned long long td0 = tile_d0, td1 = tile_d1;   // tensor dims == tile dims
  u32x8 g1;
  g1[0] = (2u << 16);                                // data_size=2 (4 bytes); mask=0
  g1[1] = (unsigned)((td0 & 0xFFFFu) << 16);         // tensor_dim0[15:0]  @ bits 63:48
  g1[2] = (unsigned)((td0 >> 16) & 0xFFFFu)          // tensor_dim0[31:16] @ bits 79:64
        | (unsigned)((td1 & 0xFFFFu) << 16);         // tensor_dim1[15:0]  @ bits 95:80
  g1[3] = (unsigned)((td1 >> 16) & 0xFFFFu)          // tensor_dim1[31:16] @ bits 111:96
        | (tile_d0 << 16);                           // tile_dim0 @ bits 127:112
  g1[4] = (tile_d1 & 0xFFFFu);                       // tile_dim1; tile_dim2=0
  g1[5] = (unsigned)(stride0 & 0xFFFFFFFFu);         // tensor_dim0_stride[31:0]
  g1[6] = (unsigned)((stride0 >> 32) & 0xFFFFu);     // tensor_dim0_stride[47:32]
  g1[7] = 0;                                         // tensor_dim1_stride (unused, 2-D)

  asm volatile("tensor_load_to_lds %0, %1" :: "s"(g0), "s"(g1) : "memory");
}

// ---------------------------------------------------------------------------
// GEMM: out[m,n] = act( sum_k A[m,k]*W[n,k] + bias[n] )
// TDM -> LDS (double-buffered, per-wave private slices, no barriers)
// -> bf16 WMMA with f32 accumulation.  K % 32 == 0.
// ---------------------------------------------------------------------------
__global__ __launch_bounds__(GWAVES * 32)
void gemm_tdm_wmma(const float* __restrict__ A, const float* __restrict__ W,
                   const float* __restrict__ bias, float* __restrict__ out,
                   int M, int N, int K, int do_relu)
{
  // [buf][wave][ A(16x32) | B(16x32) ]  = 32 KB total
  __shared__ float lds[2][GWAVES][2 * 16 * KCHUNK];

  const int lane = threadIdx.x & 31;
  const int wid  = __builtin_amdgcn_readfirstlane(threadIdx.x >> 5);
  const int half = lane >> 4;
  const int row  = lane & 15;

  const int m0 = blockIdx.x * 16;
  const int nt = blockIdx.y * GWAVES + wid;          // this wave's 16-wide N tile
  if (nt * 16 >= N) return;                          // wave-uniform exit, no barriers
  const int n0 = nt * 16;

  const float* Abase = A + (size_t)m0 * K;
  const float* Wbase = W + (size_t)n0 * K;

  float* ldsA[2] = { &lds[0][wid][0],          &lds[1][wid][0]          };
  float* ldsB[2] = { &lds[0][wid][16*KCHUNK],  &lds[1][wid][16*KCHUNK]  };
  const unsigned offA[2] = { (unsigned)(uintptr_t)(void*)ldsA[0],
                             (unsigned)(uintptr_t)(void*)ldsA[1] };
  const unsigned offB[2] = { (unsigned)(uintptr_t)(void*)ldsB[0],
                             (unsigned)(uintptr_t)(void*)ldsB[1] };

  // prime buffer 0
  tdm_load_2d(Abase, offA[0], KCHUNK, 16, (unsigned long long)K);
  tdm_load_2d(Wbase, offB[0], KCHUNK, 16, (unsigned long long)K);

  v8f acc = {};
  int cur = 0;
  for (int k0 = 0; k0 < K; k0 += KCHUNK) {
    const int more = (k0 + KCHUNK) < K;
    if (more) {                                      // issue next chunk into other buf
      tdm_load_2d(Abase + k0 + KCHUNK, offA[cur ^ 1], KCHUNK, 16, (unsigned long long)K);
      tdm_load_2d(Wbase + k0 + KCHUNK, offB[cur ^ 1], KCHUNK, 16, (unsigned long long)K);
      __builtin_amdgcn_s_wait_tensorcnt(2);          // current buffer complete
    } else {
      __builtin_amdgcn_s_wait_tensorcnt(0);          // tail: everything complete
    }
    asm volatile("" ::: "memory");

    // A frag (ISA 16-bit A 16x32 map): element e -> k = e + (e>=8?8:0) + 8*half
    const float* ap = ldsA[cur] + row * KCHUNK + 8 * half;
    // B frag (ISA 16-bit B 32x16 map): element e -> k = e + (e>=8?8:0) + 8*half, col = lane&15
    const float* bp = ldsB[cur] + row * KCHUNK + 8 * half;

    float ab[16], bb[16];
    *(float4*)(ab +  0) = *(const float4*)(ap +  0);
    *(float4*)(ab +  4) = *(const float4*)(ap +  4);
    *(float4*)(ab +  8) = *(const float4*)(ap + 16);
    *(float4*)(ab + 12) = *(const float4*)(ap + 20);
    *(float4*)(bb +  0) = *(const float4*)(bp +  0);
    *(float4*)(bb +  4) = *(const float4*)(bp +  4);
    *(float4*)(bb +  8) = *(const float4*)(bp + 16);
    *(float4*)(bb + 12) = *(const float4*)(bp + 20);

    v16bf av, bv;
#pragma unroll
    for (int e = 0; e < 16; ++e) { av[e] = (__bf16)ab[e]; bv[e] = (__bf16)bb[e]; }

    acc = __builtin_amdgcn_wmma_f32_16x16x32_bf16(
        false, av, false, bv, (short)0, acc, false, false);
    cur ^= 1;
  }

  // C/D layout: VGPR r -> M = r + 8*half, N = lane&15
#pragma unroll
  for (int r = 0; r < 8; ++r) {
    int m = m0 + r + 8 * half;
    int n = n0 + row;
    float v = acc[r] + bias[n];
    if (do_relu) v = fmaxf(v, 0.0f);
    out[(size_t)m * N + n] = v;
  }
}

// ---------------------------------------------------------------------------
// Quantum circuit: one 8-qubit statevector (256 complex amps) per wave in LDS.
// Qubit w lives at bit position (7-w) of the flattened amplitude index.
// ---------------------------------------------------------------------------
__device__ __forceinline__ float2 cmul(float2 a, float2 b) {
  return make_float2(a.x*b.x - a.y*b.y, a.x*b.y + a.y*b.x);
}
__device__ __forceinline__ float2 cadd(float2 a, float2 b) {
  return make_float2(a.x + b.x, a.y + b.y);
}
__device__ __forceinline__ int dep1(int p, int bt) {
  int low = p & ((1 << bt) - 1);
  int high = p >> bt;
  return low | (high << (bt + 1));
}
__device__ __forceinline__ int dep2(int p, int b0, int b1) {
  int lo = b0 < b1 ? b0 : b1;
  int hi = b0 < b1 ? b1 : b0;
  int low  = p & ((1 << lo) - 1);
  int mid  = (p >> lo) & ((1 << (hi - 1 - lo)) - 1);
  int high = p >> (hi - 1);
  return low | (mid << (lo + 1)) | (high << (hi + 1));
}

__device__ __forceinline__ void ctrl2x2(float2* S, int lane, int ctrl, int targ,
                                        float2 u00, float2 u01, float2 u10, float2 u11)
{
  const int bc = 7 - ctrl, bt = 7 - targ;
#pragma unroll
  for (int g = 0; g < 2; ++g) {
    int p   = lane + 32 * g;
    int i0  = dep2(p, bc, bt) | (1 << bc);
    int i1  = i0 | (1 << bt);
    float2 a = S[i0], b = S[i1];
    S[i0] = cadd(cmul(u00, a), cmul(u01, b));
    S[i1] = cadd(cmul(u10, a), cmul(u11, b));
  }
  asm volatile("" ::: "memory");
}

__device__ __forceinline__ void one2x2(float2* S, int lane, int targ,
                                       float2 u00, float2 u01, float2 u10, float2 u11)
{
  const int bt = 7 - targ;
#pragma unroll
  for (int g = 0; g < 4; ++g) {
    int p  = lane + 32 * g;
    int i0 = dep1(p, bt);
    int i1 = i0 | (1 << bt);
    float2 a = S[i0], b = S[i1];
    S[i0] = cadd(cmul(u00, a), cmul(u01, b));
    S[i1] = cadd(cmul(u10, a), cmul(u11, b));
  }
  asm volatile("" ::: "memory");
}

__global__ __launch_bounds__(256)
void quantum_kernel(const float* __restrict__ feats,   // [B, CHIPS, 8]
                    const float* __restrict__ conv,    // [CHIPS, 2, 8, 6]
                    const float* __restrict__ pool,    // [CHIPS, 2, 4, 3]
                    float* __restrict__ z)             // [B, CHIPS]
{
  __shared__ float2 smem[8][256];
  const int lane = threadIdx.x & 31;
  const int wid  = threadIdx.x >> 5;
  const int sv   = blockIdx.x * 8 + wid;
  const int b    = sv / CHIPS;
  const int c    = sv % CHIPS;
  float2* S = smem[wid];

  float cs[8], sn[8];
#pragma unroll
  for (int w = 0; w < 8; ++w) {
    float t = 0.5f * feats[(b * CHIPS + c) * 8 + w];
    cs[w] = cosf(t); sn[w] = sinf(t);
  }
#pragma unroll
  for (int g = 0; g < 8; ++g) {
    int idx = g * 32 + lane;
    float a = 1.0f;
#pragma unroll
    for (int w = 0; w < 8; ++w) a *= ((idx >> (7 - w)) & 1) ? sn[w] : cs[w];
    S[idx] = make_float2(a, 0.0f);
  }
  asm volatile("" ::: "memory");

  auto CRY = [&](int ct, int tg, float t) {
    float ch = cosf(0.5f * t), sh = sinf(0.5f * t);
    ctrl2x2(S, lane, ct, tg, make_float2(ch,0), make_float2(-sh,0),
                             make_float2(sh,0), make_float2(ch,0));
  };
  auto CRX = [&](int ct, int tg, float t) {
    float ch = cosf(0.5f * t), sh = sinf(0.5f * t);
    ctrl2x2(S, lane, ct, tg, make_float2(ch,0), make_float2(0,-sh),
                             make_float2(0,-sh), make_float2(ch,0));
  };
  auto CNOTg = [&](int ct, int tg) {
    ctrl2x2(S, lane, ct, tg, make_float2(0,0), make_float2(1,0),
                             make_float2(1,0), make_float2(0,0));
  };
  auto RYg = [&](int tg, float t) {
    float ch = cosf(0.5f * t), sh = sinf(0.5f * t);
    one2x2(S, lane, tg, make_float2(ch,0), make_float2(-sh,0),
                        make_float2(sh,0), make_float2(ch,0));
  };
  auto CU3 = [&](int ct, int tg, float t, float p, float d) {
    float ctt = cosf(0.5f * t), st = sinf(0.5f * t);
    float2 ep  = make_float2(cosf(p), sinf(p));
    float2 ed  = make_float2(cosf(d), sinf(d));
    float2 epd = cmul(ep, ed);
    ctrl2x2(S, lane, ct, tg,
            make_float2(ctt, 0),
            make_float2(-ed.x * st, -ed.y * st),
            make_float2(ep.x * st,  ep.y * st),
            make_float2(epd.x * ctt, epd.y * ctt));
  };

  // layer 0 conv on pairs (0,1)(2,3)(4,5)(6,7)
#pragma unroll
  for (int pi = 0; pi < 4; ++pi) {
    const int w1 = 2 * pi, w2 = 2 * pi + 1;
    const float* cp = conv + ((size_t)(c * QDEPTH + 0) * NQ + pi) * 6;
    float p0 = cp[0], p1 = cp[1], p2 = cp[2], p3 = cp[3], p4 = cp[4], p5 = cp[5];
    CRY(w1, w2, p0); CRY(w2, w1, p1);
    CRX(w1, w2, p2); CRX(w2, w1, p3);
    CNOTg(w1, w2);
    RYg(w1, p4); RYg(w2, p5);
  }
  // layer 0 pooling
#pragma unroll
  for (int i2 = 0; i2 < 4; ++i2) {
    const float* qp = pool + ((size_t)(c * QDEPTH + 0) * 4 + i2) * 3;
    CU3(2 * i2 + 1, 2 * i2, qp[0], qp[1], qp[2]);
  }
  // layer 1 conv on pairs (0,2)(4,6)
#pragma unroll
  for (int pi = 0; pi < 2; ++pi) {
    const int w1 = 4 * pi, w2 = 4 * pi + 2;
    const float* cp = conv + ((size_t)(c * QDEPTH + 1) * NQ + pi) * 6;
    float p0 = cp[0], p1 = cp[1], p2 = cp[2], p3 = cp[3], p4 = cp[4], p5 = cp[5];
    CRY(w1, w2, p0); CRY(w2, w1, p1);
    CRX(w1, w2, p2); CRX(w2, w1, p3);
    CNOTg(w1, w2);
    RYg(w1, p4); RYg(w2, p5);
  }
  // layer 1 pooling
#pragma unroll
  for (int i2 = 0; i2 < 2; ++i2) {
    const float* qp = pool + ((size_t)(c * QDEPTH + 1) * 4 + i2) * 3;
    CU3(4 * i2 + 2, 4 * i2, qp[0], qp[1], qp[2]);
  }

  // z = P(q0=0) - P(q0=1); q0 is the MSB of the amplitude index
  float acc = 0.0f;
#pragma unroll
  for (int g = 0; g < 8; ++g) {
    int idx = g * 32 + lane;
    float2 a = S[idx];
    float p2 = a.x * a.x + a.y * a.y;
    acc += (idx & 0x80) ? -p2 : p2;
  }
#pragma unroll
  for (int off = 16; off > 0; off >>= 1) acc += __shfl_xor(acc, off, 32);
  if (lane == 0) z[b * CHIPS + c] = acc;
}

// ---------------------------------------------------------------------------
// Head: hh = relu(z @ hw1^T + hb1); out = hh @ hw2^T + hb2   (tiny, f32)
// ---------------------------------------------------------------------------
__global__ __launch_bounds__(64)
void head_kernel(const float* __restrict__ z,   const float* __restrict__ hw1,
                 const float* __restrict__ hb1, const float* __restrict__ hw2,
                 const float* __restrict__ hb2, float* __restrict__ out)
{
  __shared__ float zr[CHIPS];
  __shared__ float hh[HID2];
  const int b = blockIdx.x, t = threadIdx.x;
  for (int k = t; k < CHIPS; k += 64) zr[k] = z[b * CHIPS + k];
  __syncthreads();
  float s = hb1[t];
  for (int k = 0; k < CHIPS; ++k) s += zr[k] * hw1[t * CHIPS + k];
  hh[t] = fmaxf(s, 0.0f);
  __syncthreads();
  if (t < ACTIONS) {
    float o = hb2[t];
#pragma unroll
    for (int j = 0; j < HID2; ++j) o += hh[j] * hw2[t * HID2 + j];
    out[b * ACTIONS + t] = o;
  }
}

// ---------------------------------------------------------------------------
extern "C" void kernel_launch(void* const* d_in, const int* in_sizes, int n_in,
                              void* d_out, int out_size, void* d_ws, size_t ws_size,
                              hipStream_t stream)
{
  const float* x    = (const float*)d_in[0];
  const float* w1   = (const float*)d_in[1];
  const float* b1   = (const float*)d_in[2];
  const float* w2   = (const float*)d_in[3];
  const float* b2   = (const float*)d_in[4];
  const float* conv = (const float*)d_in[5];
  const float* pool = (const float*)d_in[6];
  const float* hw1  = (const float*)d_in[7];
  const float* hb1  = (const float*)d_in[8];
  const float* hw2  = (const float*)d_in[9];
  const float* hb2  = (const float*)d_in[10];
  float* outF = (float*)d_out;

  float* ws    = (float*)d_ws;
  float* h     = ws;                                   // 128*256
  float* feats = h + (size_t)BATCH * HID;              // 128*800
  float* zbuf  = feats + (size_t)BATCH * CHIPS * NQ;   // 128*100

  // GEMM1: h = relu(x @ w1^T + b1)   (128 x 28224 x 256) TDM+WMMA
  gemm_tdm_wmma<<<dim3(BATCH / 16, HID / (16 * GWAVES)), GWAVES * 32, 0, stream>>>(
      x, w1, b1, h, BATCH, HID, IN_DIM, 1);
  // GEMM2: feats = h @ w2^T + b2     (128 x 256 x 800) TDM+WMMA
  gemm_tdm_wmma<<<dim3(BATCH / 16, (CHIPS * NQ / 16 + GWAVES - 1) / GWAVES),
                  GWAVES * 32, 0, stream>>>(
      h, w2, b2, feats, BATCH, CHIPS * NQ, HID, 0);
  // Quantum circuit: 12800 statevectors, 1 per wave, LDS-resident
  quantum_kernel<<<(BATCH * CHIPS) / 8, 256, 0, stream>>>(feats, conv, pool, zbuf);
  // Head MLP
  head_kernel<<<BATCH, 64, 0, stream>>>(zbuf, hw1, hb1, hw2, hb2, outF);
}
